// MixedOp_67963562492126
// MI455X (gfx1250) — matrix operand order repaired and checked by reading, not compiled, not run on hardware
//
#include <hip/hip_runtime.h>
#include <hip/hip_bf16.h>

#define NB  32
#define CCH 128
#define HH  64
#define WW  64
#define HW  4096
#define EPSV 1e-5f

typedef __attribute__((ext_vector_type(16))) __bf16 v16bf;
typedef __attribute__((ext_vector_type(8)))  float  v8f;

union FragBF { uint4 u[2]; v16bf v; };

__device__ inline v8f wmma_bf16(const FragBF& a, const FragBF& b, v8f c) {
    return __builtin_amdgcn_wmma_f32_16x16x32_bf16(
        false, a.v, false, b.v, (short)0, c, false, false);
}

// ---------------- weight f32 -> bf16 conversion (6 pointwise mats) -------------
__global__ __launch_bounds__(256)
void k_w2bf(const float* __restrict__ a, const float* __restrict__ b,
            const float* __restrict__ c, const float* __restrict__ d,
            const float* __restrict__ e, const float* __restrict__ f,
            __bf16* __restrict__ out) {
    int i = blockIdx.x * 256 + threadIdx.x;      // 6*16384 total
    int sel = i >> 14, j = i & 16383;
    const float* src;
    if      (sel == 0) src = a;
    else if (sel == 1) src = b;
    else if (sel == 2) src = c;
    else if (sel == 3) src = d;
    else if (sel == 4) src = e;
    else               src = f;
    out[i] = (__bf16)src[j];
}

// ---------------- spatial stats: mean/max over channels ------------------------
__global__ __launch_bounds__(256)
void k_spatial_stats(const float* __restrict__ x, float* __restrict__ s) {
    int idx = blockIdx.x * 256 + threadIdx.x;    // N*HW
    int n = idx >> 12, hw = idx & 4095;
    const float* p = x + (size_t)n * CCH * HW + hw;
    float sum = 0.f, mx = -1e30f;
    for (int c = 0; c < CCH; ++c) {
        float v = p[(size_t)c * HW];
        sum += v; mx = fmaxf(mx, v);
    }
    s[((size_t)n * 2)     * HW + hw] = sum * (1.f / 128.f);
    s[((size_t)n * 2 + 1) * HW + hw] = mx;
}

// ---------------- 7x7 conv (2ch -> 1ch) + sigmoid ------------------------------
__global__ __launch_bounds__(256)
void k_sa_conv(const float* __restrict__ s, const float* __restrict__ w,
               float* __restrict__ sa) {
    int idx = blockIdx.x * 256 + threadIdx.x;    // N*HW
    int n = idx >> 12, hw = idx & 4095;
    int h = hw >> 6, wx = hw & 63;
    float acc = 0.f;
    for (int ci = 0; ci < 2; ++ci) {
        const float* sp = s + ((size_t)n * 2 + ci) * HW;
        for (int kh = 0; kh < 7; ++kh) {
            int ih = h - 3 + kh;
            if ((unsigned)ih >= (unsigned)HH) continue;
            for (int kw = 0; kw < 7; ++kw) {
                int iw = wx - 3 + kw;
                if ((unsigned)iw >= (unsigned)WW) continue;
                acc += w[ci * 49 + kh * 7 + kw] * sp[ih * 64 + iw];
            }
        }
    }
    sa[idx] = 1.f / (1.f + expf(-acc));
}

// ---------------- x1 = x*sa ; per-(n,c) sum & max reductions -------------------
__global__ __launch_bounds__(256)
void k_apply_sa(const float* __restrict__ x, const float* __restrict__ sa,
                float* __restrict__ x1, float* __restrict__ chsum,
                float* __restrict__ chmax) {
    int c = blockIdx.x, n = blockIdx.y, tid = threadIdx.x;
    __shared__ float rs[256], rm[256];
    size_t plane = ((size_t)n * CCH + c) * HW;
    const float* sap = sa + (size_t)n * HW;
    float s = 0.f, mx = -1e30f;
    for (int hw = tid; hw < HW; hw += 256) {
        float v = x[plane + hw] * sap[hw];
        x1[plane + hw] = v;
        s += v; mx = fmaxf(mx, v);
    }
    rs[tid] = s; rm[tid] = mx;
    __syncthreads();
    for (int o = 128; o > 0; o >>= 1) {
        if (tid < o) { rs[tid] += rs[tid + o]; rm[tid] = fmaxf(rm[tid], rm[tid + o]); }
        __syncthreads();
    }
    if (tid == 0) { chsum[n * CCH + c] = rs[0]; chmax[n * CCH + c] = rm[0]; }
}

// ---------------- channel-attention MLP ---------------------------------------
__global__ __launch_bounds__(128)
void k_ca_mlp(const float* __restrict__ chsum, const float* __restrict__ chmax,
              const float* __restrict__ fc1, const float* __restrict__ fc2,
              float* __restrict__ att, float* __restrict__ slist) {
    __shared__ float za[128], zm[128], ha[64], hm[64];
    int n = blockIdx.x, t = threadIdx.x;
    za[t] = chsum[n * CCH + t] * (1.f / 4096.f);
    zm[t] = chmax[n * CCH + t];
    __syncthreads();
    if (t < 64) {
        float a = 0.f, m = 0.f;
        for (int c = 0; c < 128; ++c) {
            float w = fc1[t * 128 + c];
            a += w * za[c]; m += w * zm[c];
        }
        ha[t] = fmaxf(a, 0.f); hm[t] = fmaxf(m, 0.f);
    }
    __syncthreads();
    float a = 0.f, m = 0.f;
    for (int j = 0; j < 64; ++j) {
        float w = fc2[t * 64 + j];
        a += w * ha[j]; m += w * hm[j];
    }
    float sg = 1.f / (1.f + expf(-(a + m)));
    att[n * CCH + t] = sg;
    atomicAdd(&slist[t], sg);
}

// ---------------- top-k (rank sort) + num_dict + last-occurrence slots ---------
__global__ void k_topk(const float* __restrict__ slist, const int* __restrict__ perm,
                       int* __restrict__ nd, int* __restrict__ cslot) {
    __shared__ float sl[128];
    __shared__ int   nds[128];
    int t = threadIdx.x;
    sl[t] = slist[t];
    __syncthreads();
    float v = sl[t];
    int rank = 0;
    for (int j = 0; j < 128; ++j) {
        float u = sl[j];
        rank += (u > v) || (u == v && j < t);
    }
    if (rank < 96) nds[rank] = t;
    __syncthreads();
    if (t >= 96) nds[t] = perm[t - 96];
    __syncthreads();
    nd[t] = nds[t];
    int slot = -1;
    for (int j = 0; j < 128; ++j)
        if (nds[j] == t) slot = j;      // last occurrence wins
    cslot[t] = slot;
}

// ---------------- xt = (x1*att)[:, num_dict] in bf16 ---------------------------
__global__ __launch_bounds__(256)
void k_build_xt(const float* __restrict__ x1, const float* __restrict__ att,
                const int* __restrict__ nd, __bf16* __restrict__ xt) {
    int idx = blockIdx.x * 256 + threadIdx.x;
    int n = idx >> 19, j = (idx >> 12) & 127, hw = idx & 4095;
    int cd = nd[j];
    float a = att[n * CCH + cd];
    xt[idx] = (__bf16)(x1[(((size_t)n * CCH + cd) << 12) + hw] * a);
}

// ---------------- 3x3 pool helpers ---------------------------------------------
__device__ inline void pool3(const __bf16* __restrict__ plane, int h, int w,
                             float& mx, float& av) {
    int h0 = max(h - 1, 0), h1 = min(h + 1, HH - 1);
    int w0 = max(w - 1, 0), w1 = min(w + 1, WW - 1);
    float m = -1e30f, s = 0.f;
    for (int ih = h0; ih <= h1; ++ih)
        for (int iw = w0; iw <= w1; ++iw) {
            float v = (float)plane[ih * 64 + iw];
            m = fmaxf(m, v); s += v;
        }
    mx = m;
    av = s / (float)((h1 - h0 + 1) * (w1 - w0 + 1));
}

__global__ __launch_bounds__(256)
void k_pool_stats(const __bf16* __restrict__ xt, float* __restrict__ stats) {
    int chunk = blockIdx.x, slot = blockIdx.y, n = blockIdx.z, tid = threadIdx.x;
    const __bf16* plane = xt + ((size_t)n * CCH + slot) * HW;
    float sm = 0.f, qm = 0.f, sa = 0.f, qa = 0.f;
    for (int e = 0; e < 4; ++e) {
        int hw = chunk * 1024 + tid * 4 + e;
        int h = hw >> 6, w = hw & 63;
        float mx, av; pool3(plane, h, w, mx, av);
        sm += mx; qm += mx * mx; sa += av; qa += av * av;
    }
    __shared__ float r0[256], r1[256], r2[256], r3[256];
    r0[tid] = sm; r1[tid] = qm; r2[tid] = sa; r3[tid] = qa;
    __syncthreads();
    for (int o = 128; o > 0; o >>= 1) {
        if (tid < o) {
            r0[tid] += r0[tid + o]; r1[tid] += r1[tid + o];
            r2[tid] += r2[tid + o]; r3[tid] += r3[tid + o];
        }
        __syncthreads();
    }
    if (tid == 0) {
        atomicAdd(&stats[(0 * CCH + slot) * 2 + 0], r0[0]);
        atomicAdd(&stats[(0 * CCH + slot) * 2 + 1], r1[0]);
        atomicAdd(&stats[(1 * CCH + slot) * 2 + 0], r2[0]);
        atomicAdd(&stats[(1 * CCH + slot) * 2 + 1], r3[0]);
    }
}

__global__ __launch_bounds__(256)
void k_pool_apply(const __bf16* __restrict__ xt, const float* __restrict__ stats,
                  const float* __restrict__ wts, float* __restrict__ temp1) {
    int chunk = blockIdx.x, slot = blockIdx.y, n = blockIdx.z, tid = threadIdx.x;
    size_t plane = ((size_t)n * CCH + slot) * HW;
    const __bf16* xp = xt + plane;
    const float cnt = (float)(NB * HW);
    float m0 = stats[(0 * CCH + slot) * 2] / cnt;
    float v0 = stats[(0 * CCH + slot) * 2 + 1] / cnt - m0 * m0;
    float i0 = rsqrtf(v0 + EPSV);
    float m1 = stats[(1 * CCH + slot) * 2] / cnt;
    float v1 = stats[(1 * CCH + slot) * 2 + 1] / cnt - m1 * m1;
    float i1 = rsqrtf(v1 + EPSV);
    float w1 = wts[1], w2 = wts[2], w3 = wts[3];
    for (int e = 0; e < 4; ++e) {
        int hw = chunk * 1024 + tid * 4 + e;
        int h = hw >> 6, w = hw & 63;
        float mx, av; pool3(xp, h, w, mx, av);
        temp1[plane + hw] = w3 * (float)xp[hw]
                          + w1 * (mx - m0) * i0
                          + w2 * (av - m1) * i1;
    }
}

// ---------------- depthwise conv (relu on load, optional bn on load) -----------
__global__ __launch_bounds__(256)
void k_dwconv(const __bf16* __restrict__ inb, const float* __restrict__ inf,
              const float* __restrict__ st, const float* __restrict__ dww,
              int ksz, int dil, __bf16* __restrict__ out) {
    __shared__ float wsh[25];
    __shared__ float sMean, sRstd;
    int chunk = blockIdx.x, slot = blockIdx.y, n = blockIdx.z, tid = threadIdx.x;
    if (tid < ksz * ksz) wsh[tid] = dww[slot * ksz * ksz + tid];
    if (tid == 0) {
        if (st) {
            const float cnt = (float)(NB * HW);
            float m = st[slot * 2] / cnt;
            float v = st[slot * 2 + 1] / cnt - m * m;
            sMean = m; sRstd = rsqrtf(v + EPSV);
        } else { sMean = 0.f; sRstd = 1.f; }
    }
    __syncthreads();
    int pad = (ksz >> 1) * dil;
    size_t plane = ((size_t)n * CCH + slot) * HW;
    float mean = sMean, rstd = sRstd;
    for (int e = 0; e < 4; ++e) {
        int hw = chunk * 1024 + tid * 4 + e;
        int h = hw >> 6, w = hw & 63;
        float acc = 0.f;
        for (int kh = 0; kh < ksz; ++kh) {
            int ih = h - pad + kh * dil;
            if ((unsigned)ih >= (unsigned)HH) continue;
            for (int kw = 0; kw < ksz; ++kw) {
                int iw = w - pad + kw * dil;
                if ((unsigned)iw >= (unsigned)WW) continue;
                float v;
                if (st) {
                    v = inf[plane + ih * 64 + iw];
                    v = (v - mean) * rstd;
                } else {
                    v = (float)inb[plane + ih * 64 + iw];
                }
                v = fmaxf(v, 0.f);
                acc += wsh[kh * ksz + kw] * v;
            }
        }
        out[plane + hw] = (__bf16)acc;
    }
}

// ---------------- WMMA bf16 pointwise conv: Y[128,4096] = W[128,128] x X --------
// Block: 256 threads = 8 waves = 4 M-groups (32 rows each) x 2 N-halves (64 px).
// Each wave register-blocks 2 M-tiles so every B fragment feeds 2 WMMAs.
// LDS tile: 128 pixels x 136 halves (pixel-major) -> B fragments are two K-contiguous
// 16B runs per lane 32B apart, 16B-aligned ds_load_b128 pairs.
__global__ __launch_bounds__(256)
void k_gemm_bn(const __bf16* __restrict__ X, const __bf16* __restrict__ Wm,
               float* __restrict__ Y, float* __restrict__ st) {
    __shared__ __bf16 lds[128 * 136];
    const int pb  = blockIdx.x;                  // pixel block (0..31), 128 px
    const int n   = blockIdx.y;
    const int tid = threadIdx.x;
    const __bf16* base = X + (size_t)n * CCH * HW + pb * 128;
    union U8 { uint4 u; __bf16 h[8]; };
    // transpose-load: global [K=128][128 px] -> LDS [pixel][K], b128 global reads
    for (int idx = tid; idx < 128 * 16; idx += 256) {
        int k = idx >> 4, pg = (idx & 15) << 3;
        U8 v; v.u = *(const uint4*)(base + (size_t)k * HW + pg);
#pragma unroll
        for (int e = 0; e < 8; ++e)
            lds[(pg + e) * 136 + k] = v.h[e];
    }
    __syncthreads();

    const int wv   = tid >> 5;
    const int mg   = wv & 3;                     // M-group: rows 32*mg..+31
    const int ng   = wv >> 2;                    // pixel half: 64*ng
    const int lane = tid & 31;
    const int lr   = lane & 15;
    const int hi   = lane >> 4;
    const int klo  = hi ? 8 : 0;                 // 16-bit A fragment K layout
    const int khlf = hi ? 8 : 0;                 // 16-bit B fragment K layout

    FragBF A[2][4];
#pragma unroll
    for (int m = 0; m < 2; ++m)
#pragma unroll
        for (int kb = 0; kb < 4; ++kb) {
            const __bf16* wp = Wm + ((mg * 2 + m) * 16 + lr) * 128 + kb * 32 + klo;
            A[m][kb].u[0] = *(const uint4*)wp;
            A[m][kb].u[1] = *(const uint4*)(wp + 16);
        }

    const v8f vzero = {0.f, 0.f, 0.f, 0.f, 0.f, 0.f, 0.f, 0.f};
    v8f acc[2][4];
#pragma unroll
    for (int m = 0; m < 2; ++m)
#pragma unroll
        for (int nt = 0; nt < 4; ++nt) acc[m][nt] = vzero;

    for (int nt = 0; nt < 4; ++nt) {
#pragma unroll
        for (int kb = 0; kb < 4; ++kb) {
            FragBF B;
            const __bf16* bp = lds + (ng * 64 + nt * 16 + lr) * 136 + kb * 32 + khlf;
            B.u[0] = *(const uint4*)bp;
            B.u[1] = *(const uint4*)(bp + 16);
            acc[0][nt] = wmma_bf16(A[0][kb], B, acc[0][nt]);
            acc[1][nt] = wmma_bf16(A[1][kb], B, acc[1][nt]);
        }
    }

    // stores + bn stats (lane reduction hoisted out of the nt loop)
#pragma unroll
    for (int m = 0; m < 2; ++m) {
#pragma unroll
        for (int r = 0; r < 8; ++r) {
            int row = (mg * 2 + m) * 16 + r + hi * 8;
            float s = 0.f, q = 0.f;
#pragma unroll
            for (int nt = 0; nt < 4; ++nt) {
                float v = acc[m][nt][r];
                int col = pb * 128 + ng * 64 + nt * 16 + lr;
                Y[((size_t)n * CCH + row) * HW + col] = v;
                s += v; q += v * v;
            }
            s += __shfl_xor(s, 1, 32); q += __shfl_xor(q, 1, 32);
            s += __shfl_xor(s, 2, 32); q += __shfl_xor(q, 2, 32);
            s += __shfl_xor(s, 4, 32); q += __shfl_xor(q, 4, 32);
            s += __shfl_xor(s, 8, 32); q += __shfl_xor(q, 8, 32);
            if (lr == 0) {
                atomicAdd(&st[row * 2 + 0], s);
                atomicAdd(&st[row * 2 + 1], q);
            }
        }
    }
}

// ---------------- temp1 += w * bn(gemm_out) ------------------------------------
__global__ __launch_bounds__(256)
void k_accum_bn(const float* __restrict__ g, const float* __restrict__ st,
                const float* __restrict__ wts, int widx, float* __restrict__ temp1) {
    int idx = blockIdx.x * 256 + threadIdx.x;
    int slot = (idx >> 12) & 127;
    const float cnt = (float)(NB * HW);
    float m = st[slot * 2] / cnt;
    float v = st[slot * 2 + 1] / cnt - m * m;
    float r = rsqrtf(v + EPSV);
    temp1[idx] += wts[widx] * (g[idx] - m) * r;
}

// ---------------- final: scatter temp1 back through num_dict -------------------
__global__ __launch_bounds__(256)
void k_scatter(const float* __restrict__ x1, const float* __restrict__ att,
               const int* __restrict__ cslot, const float* __restrict__ temp1,
               float* __restrict__ out) {
    int idx = blockIdx.x * 256 + threadIdx.x;
    int n = idx >> 19, c = (idx >> 12) & 127, hw = idx & 4095;
    int s = cslot[c];
    out[idx] = (s >= 0) ? temp1[(((size_t)n * CCH + s) << 12) + hw]
                        : x1[idx] * att[n * CCH + c];
}

// ================================================================================
extern "C" void kernel_launch(void* const* d_in, const int* in_sizes, int n_in,
                              void* d_out, int out_size, void* d_ws, size_t ws_size,
                              hipStream_t stream) {
    (void)in_sizes; (void)n_in; (void)out_size; (void)ws_size;
    const float* x        = (const float*)d_in[0];
    const float* wts      = (const float*)d_in[1];
    const float* ca_fc1   = (const float*)d_in[2];
    const float* ca_fc2   = (const float*)d_in[3];
    const float* sa_w     = (const float*)d_in[4];
    const float* sep3_dw1 = (const float*)d_in[5];
    const float* sep3_pw1 = (const float*)d_in[6];
    const float* sep3_dw2 = (const float*)d_in[7];
    const float* sep3_pw2 = (const float*)d_in[8];
    const float* sep5_dw1 = (const float*)d_in[9];
    const float* sep5_pw1 = (const float*)d_in[10];
    const float* sep5_dw2 = (const float*)d_in[11];
    const float* sep5_pw2 = (const float*)d_in[12];
    const float* dil3_dw  = (const float*)d_in[13];
    const float* dil3_pw  = (const float*)d_in[14];
    const float* dil5_dw  = (const float*)d_in[15];
    const float* dil5_pw  = (const float*)d_in[16];
    const int*   perm     = (const int*)d_in[17];

    char* wbase = (char*)d_ws;
    size_t o = 0;
    auto alloc = [&](size_t bytes) -> void* {
        void* p = wbase + o;
        o += (bytes + 255) & ~(size_t)255;
        return p;
    };
    const size_t TENS = (size_t)NB * CCH * HW;
    float*  x1    = (float*)alloc(sizeof(float) * TENS);
    __bf16* xt    = (__bf16*)alloc(2 * TENS);
    __bf16* act   = (__bf16*)alloc(2 * TENS);
    float*  gbuf  = (float*)alloc(sizeof(float) * TENS);
    float*  temp1 = (float*)alloc(sizeof(float) * TENS);
    float*  sbuf  = (float*)alloc(sizeof(float) * NB * 2 * HW);
    float*  samap = (float*)alloc(sizeof(float) * NB * HW);
    float*  chsum = (float*)alloc(sizeof(float) * NB * CCH);
    float*  chmax = (float*)alloc(sizeof(float) * NB * CCH);
    float*  att   = (float*)alloc(sizeof(float) * NB * CCH);
    float*  slist = (float*)alloc(sizeof(float) * CCH);
    int*    nd    = (int*)alloc(sizeof(int) * CCH);
    int*    cslot = (int*)alloc(sizeof(int) * CCH);
    float*  stats = (float*)alloc(sizeof(float) * 8 * CCH * 2);
    __bf16* wbf   = (__bf16*)alloc(2ull * 6 * CCH * CCH);

    hipMemsetAsync(slist, 0, sizeof(float) * CCH, stream);
    hipMemsetAsync(stats, 0, sizeof(float) * 8 * CCH * 2, stream);

    const int PIX_BLKS = NB * HW / 256;          // 512
    const int ELT_BLKS = (int)(TENS / 256);      // 65536
    dim3 planeGrid(4, CCH, NB);
    dim3 gemmGrid(32, NB);                       // 128 px per block

    k_w2bf<<<6 * 16384 / 256, 256, 0, stream>>>(sep3_pw1, sep3_pw2, sep5_pw1,
                                                sep5_pw2, dil3_pw, dil5_pw, wbf);
    k_spatial_stats<<<PIX_BLKS, 256, 0, stream>>>(x, sbuf);
    k_sa_conv<<<PIX_BLKS, 256, 0, stream>>>(sbuf, sa_w, samap);
    k_apply_sa<<<dim3(CCH, NB), 256, 0, stream>>>(x, samap, x1, chsum, chmax);
    k_ca_mlp<<<NB, 128, 0, stream>>>(chsum, chmax, ca_fc1, ca_fc2, att, slist);
    k_topk<<<1, 128, 0, stream>>>(slist, perm, nd, cslot);
    k_build_xt<<<ELT_BLKS, 256, 0, stream>>>(x1, att, nd, xt);

    // pools + identity term -> temp1
    k_pool_stats<<<planeGrid, 256, 0, stream>>>(xt, stats);
    k_pool_apply<<<planeGrid, 256, 0, stream>>>(xt, stats, wts, temp1);

    // sep3: relu->dw3->pw1->bn->relu->dw3->pw2->bn, weight[4]
    k_dwconv<<<planeGrid, 256, 0, stream>>>(xt, nullptr, nullptr, sep3_dw1, 3, 1, act);
    k_gemm_bn<<<gemmGrid, 256, 0, stream>>>(act, wbf + 0 * 16384, gbuf, stats + 2 * CCH * 2);
    k_dwconv<<<planeGrid, 256, 0, stream>>>(nullptr, gbuf, stats + 2 * CCH * 2, sep3_dw2, 3, 1, act);
    k_gemm_bn<<<gemmGrid, 256, 0, stream>>>(act, wbf + 1 * 16384, gbuf, stats + 3 * CCH * 2);
    k_accum_bn<<<ELT_BLKS, 256, 0, stream>>>(gbuf, stats + 3 * CCH * 2, wts, 4, temp1);

    // sep5, weight[5]
    k_dwconv<<<planeGrid, 256, 0, stream>>>(xt, nullptr, nullptr, sep5_dw1, 5, 1, act);
    k_gemm_bn<<<gemmGrid, 256, 0, stream>>>(act, wbf + 2 * 16384, gbuf, stats + 4 * CCH * 2);
    k_dwconv<<<planeGrid, 256, 0, stream>>>(nullptr, gbuf, stats + 4 * CCH * 2, sep5_dw2, 5, 1, act);
    k_gemm_bn<<<gemmGrid, 256, 0, stream>>>(act, wbf + 3 * 16384, gbuf, stats + 5 * CCH * 2);
    k_accum_bn<<<ELT_BLKS, 256, 0, stream>>>(gbuf, stats + 5 * CCH * 2, wts, 5, temp1);

    // dil3 (dilation 2, pad 2), weight[6]
    k_dwconv<<<planeGrid, 256, 0, stream>>>(xt, nullptr, nullptr, dil3_dw, 3, 2, act);
    k_gemm_bn<<<gemmGrid, 256, 0, stream>>>(act, wbf + 4 * 16384, gbuf, stats + 6 * CCH * 2);
    k_accum_bn<<<ELT_BLKS, 256, 0, stream>>>(gbuf, stats + 6 * CCH * 2, wts, 6, temp1);

    // dil5 (dilation 2, pad 4), weight[7]
    k_dwconv<<<planeGrid, 256, 0, stream>>>(xt, nullptr, nullptr, dil5_dw, 5, 2, act);
    k_gemm_bn<<<gemmGrid, 256, 0, stream>>>(act, wbf + 5 * 16384, gbuf, stats + 7 * CCH * 2);
    k_accum_bn<<<ELT_BLKS, 256, 0, stream>>>(gbuf, stats + 7 * CCH * 2, wts, 7, temp1);

    k_scatter<<<ELT_BLKS, 256, 0, stream>>>(x1, att, cslot, temp1, (float*)d_out);
}
